// _Expert_quant_noise_76476187673237
// MI455X (gfx1250) — matrix operand order repaired and checked by reading, not compiled, not run on hardware
//
#include <hip/hip_runtime.h>
#include <stdint.h>

// ---------------------------------------------------------------------------
// MoE FFN (8 experts x 2048 tokens, D=1024, H=4096), balanced load.
//   k1: act = gelu(inp @ W1^T + b1)   -> bf16 in d_ws  [T, H]
//   k2: out = act @ W2^T + b2         -> f32 in d_out  [T, D]
// bf16 WMMA (v_wmma_f32_16x16x32_bf16), f32 accumulate; 275 GFLOP, ~0.4 GB HBM
// -> math-bound, so maximize WMMA rate: 64x64 wave tiles (16 WMMA : 16 ds_load
// per K-step), double-buffered LDS, async global->LDS staging in kernel 2,
// branch-free sigmoid-form GELU (1 v_exp + 1 v_rcp per element).
// ---------------------------------------------------------------------------

typedef __bf16    v16bf __attribute__((ext_vector_type(16)));
typedef float     v8f   __attribute__((ext_vector_type(8)));
typedef float     f32x4 __attribute__((ext_vector_type(4)));
typedef uint32_t  u32x4 __attribute__((ext_vector_type(4)));
typedef int       i32x4 __attribute__((ext_vector_type(4)));

constexpr int T_TOK = 16384;
constexpr int NE    = 8;
constexpr int DM    = 1024;       // d_model
constexpr int DH    = 4096;       // d_hidden
constexpr int CAP   = T_TOK / NE; // 2048 tokens per expert (balanced)

constexpr int BM = 128, BN = 256, BK = 32;   // workgroup tile / K tile
constexpr int MI = 4, NI = 4;                // 16x16 sub-tiles per wave (64x64)
constexpr int LDS_STRIDE = 80;               // 64B K-slice + 16B pad (conflict-free)

#if defined(__has_builtin)
#  if __has_builtin(__builtin_amdgcn_global_load_async_to_lds_b128)
#    define MOE_ASYNC_COPY 1
#  endif
#endif
#ifndef MOE_ASYNC_COPY
#  define MOE_ASYNC_COPY 0
#endif

// Builtin prototype (from hipcc diagnostic): (int4 AS1*, int4 AS3*, imm, imm).
typedef __attribute__((address_space(1))) i32x4* gv4_p;
typedef __attribute__((address_space(3))) i32x4* lv4_p;

__device__ __forceinline__ uint32_t pack2_bf16(float x, float y) {
  uint32_t a = __builtin_bit_cast(uint32_t, x);
  uint32_t b = __builtin_bit_cast(uint32_t, y);
  a = (a + 0x7FFFu + ((a >> 16) & 1u)) >> 16;            // RNE, low half
  b = (b + 0x7FFFu + ((b >> 16) & 1u)) & 0xFFFF0000u;    // RNE, high half
  return a | b;
}

__device__ __forceinline__ unsigned short f32_to_bf16(float x) {
  uint32_t u = __builtin_bit_cast(uint32_t, x);
  u = (u + 0x7FFFu + ((u >> 16) & 1u)) >> 16;
  return (unsigned short)u;
}

__device__ __forceinline__ float fast_exp2(float x) {
#if __has_builtin(__builtin_amdgcn_exp2f)
  return __builtin_amdgcn_exp2f(x);
#else
  return exp2f(x);
#endif
}

__device__ __forceinline__ float fast_rcp(float x) {
#if __has_builtin(__builtin_amdgcn_rcpf)
  return __builtin_amdgcn_rcpf(x);
#else
  return 1.0f / x;
#endif
}

// gelu_tanh(x) = x * sigmoid(2*sqrt(2/pi)*(x + 0.044715 x^3)); branch-free.
// exp2(-C*u): +inf -> rcp -> 0 (x<<0), 0 -> rcp(1) = 1 (x>>0). No NaN paths.
__device__ __forceinline__ float gelu_fast(float x) {
  const float C = 2.3022082f;   // 2*sqrt(2/pi)*log2(e)
  float u = x + 0.044715f * x * x * x;
  float ex = fast_exp2(-C * u);
  return x * fast_rcp(1.0f + ex);
}

// --- LDS staging -----------------------------------------------------------
// 128-row x 32-K fp32 tile -> bf16 LDS tile; 256 threads = (row, half).
__device__ __forceinline__ void stage_tile_f32_128(uint8_t* lds, const float* __restrict__ src,
                                                   size_t ld, int tid) {
  const int row = tid >> 1, half = tid & 1;
  const f32x4* p = (const f32x4*)(src + (size_t)row * ld + half * 16);
  f32x4 f0 = p[0], f1 = p[1], f2 = p[2], f3 = p[3];
  u32x4 a, b;
  a[0] = pack2_bf16(f0[0], f0[1]);  a[1] = pack2_bf16(f0[2], f0[3]);
  a[2] = pack2_bf16(f1[0], f1[1]);  a[3] = pack2_bf16(f1[2], f1[3]);
  b[0] = pack2_bf16(f2[0], f2[1]);  b[1] = pack2_bf16(f2[2], f2[3]);
  b[2] = pack2_bf16(f3[0], f3[1]);  b[3] = pack2_bf16(f3[2], f3[3]);
  uint8_t* dst = lds + row * LDS_STRIDE + half * 32;
  *(u32x4*)(dst)      = a;
  *(u32x4*)(dst + 16) = b;
}

// 256-row x 32-K fp32 tile -> bf16 LDS tile; 256 threads = one full row each.
__device__ __forceinline__ void stage_tile_f32_256(uint8_t* lds, const float* __restrict__ src,
                                                   size_t ld, int tid) {
  const int row = tid;
  const f32x4* p = (const f32x4*)(src + (size_t)row * ld);
  f32x4 f[8];
#pragma unroll
  for (int j = 0; j < 8; ++j) f[j] = p[j];
  uint8_t* dst = lds + row * LDS_STRIDE;
#pragma unroll
  for (int j = 0; j < 4; ++j) {
    u32x4 o;
    o[0] = pack2_bf16(f[2 * j][0], f[2 * j][1]);
    o[1] = pack2_bf16(f[2 * j][2], f[2 * j][3]);
    o[2] = pack2_bf16(f[2 * j + 1][0], f[2 * j + 1][1]);
    o[3] = pack2_bf16(f[2 * j + 1][2], f[2 * j + 1][3]);
    *(u32x4*)(dst + 16 * j) = o;
  }
}

// 128-row x 32-K bf16 tile, already bf16 in global -> async copy to LDS.
__device__ __forceinline__ void stage_tile_bf16_128(uint8_t* lds, const unsigned short* __restrict__ src,
                                                    size_t ld, int tid) {
  const int row = tid >> 1, half = tid & 1;
  const uint8_t* gp = (const uint8_t*)(src + (size_t)row * ld + half * 16);
  uint8_t*       lp = lds + row * LDS_STRIDE + half * 32;
#if MOE_ASYNC_COPY
  uint8_t* gnc = (uint8_t*)gp;   // strip const in generic AS first
  __builtin_amdgcn_global_load_async_to_lds_b128((gv4_p)gnc,        (lv4_p)lp,        0, 0);
  __builtin_amdgcn_global_load_async_to_lds_b128((gv4_p)(gnc + 16), (lv4_p)(lp + 16), 0, 0);
#else
  *(u32x4*)(lp)      = *(const u32x4*)(gp);
  *(u32x4*)(lp + 16) = *(const u32x4*)(gp + 16);
#endif
}

__device__ __forceinline__ void wait_async_then_barrier() {
#if MOE_ASYNC_COPY
#  if __has_builtin(__builtin_amdgcn_s_wait_asynccnt)
  __builtin_amdgcn_s_wait_asynccnt(0);
#  else
  asm volatile("s_wait_asynccnt 0" ::: "memory");
#  endif
#endif
  __syncthreads();
}

// --- WMMA fragment loads (ISA 7.12.2 layouts) ------------------------------
// A (16x32 bf16): lane m<16 -> row m, K 0-7 & 16-23 (bytes +0,+32);
//                 lane m+16 -> row m, K 8-15 & 24-31 (bytes +16,+48).
__device__ __forceinline__ v16bf load_fragA(const uint8_t* tile, int rowBlock, int lane) {
  const int r  = rowBlock * 16 + (lane & 15);
  const int c0 = (lane >> 4) << 4;
  const uint8_t* p = tile + r * LDS_STRIDE + c0;
  union { u32x4 q[2]; v16bf v; } u;
  u.q[0] = *(const u32x4*)(p);
  u.q[1] = *(const u32x4*)(p + 32);
  return u.v;
}

// B (32x16 bf16): lane n<16 -> column n, K 0-7 & 16-23 (bytes +0,+32);
//                 lane n+16 -> column n, K 8-15 & 24-31 (bytes +16,+48).
__device__ __forceinline__ v16bf load_fragB(const uint8_t* tile, int rowBlock, int lane) {
  const int n = rowBlock * 16 + (lane & 15);
  const uint8_t* p = tile + n * LDS_STRIDE + ((lane >> 4) << 4);
  union { u32x4 q[2]; v16bf v; } u;
  u.q[0] = *(const u32x4*)(p);
  u.q[1] = *(const u32x4*)(p + 32);
  return u.v;
}

// ---------------------------------------------------------------------------
// Kernel 1: act[t,h] = gelu( sum_d inp[t,d]*W1[e,h,d] + b1[e,h] ), bf16 out.
// grid = (DH/BN, CAP/BM, NE), block = 256 (8 waves: 2 M-waves x 4 N-waves).
// ---------------------------------------------------------------------------
__global__ __launch_bounds__(256) void moe_ffn1(const float* __restrict__ inp,
                                                const float* __restrict__ W1,
                                                const float* __restrict__ b1,
                                                unsigned short* __restrict__ act) {
  __shared__ __align__(16) uint8_t ldsA[2][BM * LDS_STRIDE];
  __shared__ __align__(16) uint8_t ldsB[2][BN * LDS_STRIDE];

  const int tid  = threadIdx.x;
  const int lane = tid & 31, wid = tid >> 5;
  const int waveM = wid & 1, waveN = wid >> 1;
  const int e = blockIdx.z, bm = blockIdx.y, bn = blockIdx.x;

  const float* Abase = inp + ((size_t)e * CAP + (size_t)bm * BM) * DM;
  const float* Bbase = W1  + ((size_t)e * DH  + (size_t)bn * BN) * DM;

  v8f acc[MI][NI] = {};

  stage_tile_f32_128(ldsA[0], Abase, DM, tid);
  stage_tile_f32_256(ldsB[0], Bbase, DM, tid);
  __syncthreads();

  for (int k0 = 0; k0 < DM; k0 += BK) {
    const int cur = (k0 / BK) & 1, nxt = cur ^ 1;
    if (k0 + BK < DM) {                          // prefetch next K tile
      stage_tile_f32_128(ldsA[nxt], Abase + k0 + BK, DM, tid);
      stage_tile_f32_256(ldsB[nxt], Bbase + k0 + BK, DM, tid);
    }
    v16bf afr[MI], bfr[NI];
#pragma unroll
    for (int mi = 0; mi < MI; ++mi) afr[mi] = load_fragA(ldsA[cur], waveM * MI + mi, lane);
#pragma unroll
    for (int ni = 0; ni < NI; ++ni) bfr[ni] = load_fragB(ldsB[cur], waveN * NI + ni, lane);
#pragma unroll
    for (int mi = 0; mi < MI; ++mi)
#pragma unroll
      for (int ni = 0; ni < NI; ++ni)
        acc[mi][ni] = __builtin_amdgcn_wmma_f32_16x16x32_bf16(
            false, afr[mi], false, bfr[ni], (short)0, acc[mi][ni], false, false);
    __syncthreads();
  }

  // Epilogue: C/D layout -> lane = N, VGPR i = M (i or i+8 by lane half).
  const int lane15 = lane & 15, laneh = lane >> 4;
  float bias[NI];
#pragma unroll
  for (int ni = 0; ni < NI; ++ni)
    bias[ni] = b1[(size_t)e * DH + bn * BN + waveN * 64 + ni * 16 + lane15];
#pragma unroll
  for (int mi = 0; mi < MI; ++mi) {
#pragma unroll
    for (int ni = 0; ni < NI; ++ni) {
      const int col = bn * BN + waveN * 64 + ni * 16 + lane15;
#pragma unroll
      for (int i = 0; i < 8; ++i) {
        const int row = e * CAP + bm * BM + waveM * 64 + mi * 16 + laneh * 8 + i;
        act[(size_t)row * DH + col] = f32_to_bf16(gelu_fast(acc[mi][ni][i] + bias[ni]));
      }
    }
  }
}

// ---------------------------------------------------------------------------
// Kernel 2: out[t,d] = sum_h act[t,h]*W2[e,d,h] + b2[e,d], f32 out.
// grid = (DM/BN, CAP/BM, NE), block = 256.
// ---------------------------------------------------------------------------
__global__ __launch_bounds__(256) void moe_ffn2(const unsigned short* __restrict__ act,
                                                const float* __restrict__ W2,
                                                const float* __restrict__ b2,
                                                float* __restrict__ out) {
  __shared__ __align__(16) uint8_t ldsA[2][BM * LDS_STRIDE];
  __shared__ __align__(16) uint8_t ldsB[2][BN * LDS_STRIDE];

  const int tid  = threadIdx.x;
  const int lane = tid & 31, wid = tid >> 5;
  const int waveM = wid & 1, waveN = wid >> 1;
  const int e = blockIdx.z, bm = blockIdx.y, bn = blockIdx.x;

  const unsigned short* Abase = act + ((size_t)e * CAP + (size_t)bm * BM) * DH;
  const float*          Bbase = W2  + ((size_t)e * DM  + (size_t)bn * BN) * DH;

  v8f acc[MI][NI] = {};

  stage_tile_bf16_128(ldsA[0], Abase, DH, tid);   // async global->LDS
  stage_tile_f32_256 (ldsB[0], Bbase, DH, tid);
  wait_async_then_barrier();

  for (int k0 = 0; k0 < DH; k0 += BK) {
    const int cur = (k0 / BK) & 1, nxt = cur ^ 1;
    if (k0 + BK < DH) {
      stage_tile_bf16_128(ldsA[nxt], Abase + k0 + BK, DH, tid);
      stage_tile_f32_256 (ldsB[nxt], Bbase + k0 + BK, DH, tid);
    }
    v16bf afr[MI], bfr[NI];
#pragma unroll
    for (int mi = 0; mi < MI; ++mi) afr[mi] = load_fragA(ldsA[cur], waveM * MI + mi, lane);
#pragma unroll
    for (int ni = 0; ni < NI; ++ni) bfr[ni] = load_fragB(ldsB[cur], waveN * NI + ni, lane);
#pragma unroll
    for (int mi = 0; mi < MI; ++mi)
#pragma unroll
      for (int ni = 0; ni < NI; ++ni)
        acc[mi][ni] = __builtin_amdgcn_wmma_f32_16x16x32_bf16(
            false, afr[mi], false, bfr[ni], (short)0, acc[mi][ni], false, false);
    wait_async_then_barrier();
  }

  const int lane15 = lane & 15, laneh = lane >> 4;
  float bias[NI];
#pragma unroll
  for (int ni = 0; ni < NI; ++ni)
    bias[ni] = b2[(size_t)e * DM + bn * BN + waveN * 64 + ni * 16 + lane15];
#pragma unroll
  for (int mi = 0; mi < MI; ++mi) {
#pragma unroll
    for (int ni = 0; ni < NI; ++ni) {
      const int col = bn * BN + waveN * 64 + ni * 16 + lane15;
#pragma unroll
      for (int i = 0; i < 8; ++i) {
        const int row = e * CAP + bm * BM + waveM * 64 + mi * 16 + laneh * 8 + i;
        out[(size_t)row * DM + col] = acc[mi][ni][i] + bias[ni];
      }
    }
  }
}

// ---------------------------------------------------------------------------
// Inputs (setup_inputs order): 0=inp f32[T,D], 1=fwd_expert_count i32[E] (balanced,
// statically T/E), 2=W1 f32[E,H,D], 3=b1 f32[E,H], 4=W2 f32[E,D,H], 5=b2 f32[E,D].
// d_ws: bf16 activations [T, H] = 128 MB.
// ---------------------------------------------------------------------------
extern "C" void kernel_launch(void* const* d_in, const int* in_sizes, int n_in,
                              void* d_out, int out_size, void* d_ws, size_t ws_size,
                              hipStream_t stream) {
  (void)in_sizes; (void)n_in; (void)out_size; (void)ws_size;
  const float* inp = (const float*)d_in[0];
  const float* W1  = (const float*)d_in[2];
  const float* b1  = (const float*)d_in[3];
  const float* W2  = (const float*)d_in[4];
  const float* b2  = (const float*)d_in[5];
  unsigned short* actb = (unsigned short*)d_ws;
  float* out = (float*)d_out;

  dim3 block(256);
  dim3 g1(DH / BN, CAP / BM, NE);   // (16, 16, 8)
  dim3 g2(DM / BN, CAP / BM, NE);   // ( 4, 16, 8)
  moe_ffn1<<<g1, block, 0, stream>>>(inp, W1, b1, actb);
  moe_ffn2<<<g2, block, 0, stream>>>(actb, W2, b2, out);
}
